// MoELayer_13219909337401
// MI455X (gfx1250) — matrix operand order repaired and checked by reading, not compiled, not run on hardware
//
#include <hip/hip_runtime.h>
#include <hip/hip_bf16.h>
#include <stdint.h>

// ---------------------------------------------------------------------------
// MoE (T=8192, D=1024, E=8, F=2048, top-2) for gfx1250 / MI455X.
// bf16 WMMA (v_wmma_f32_16x16x32_bf16), fp32 accum, double-buffered LDS,
// async global->LDS DMA (ASYNCcnt) for the bf16 H tiles in the down-proj.
// ---------------------------------------------------------------------------

typedef __attribute__((ext_vector_type(16))) __bf16 v16bf;
typedef __attribute__((ext_vector_type(8)))  __bf16 v8bf;
typedef __attribute__((ext_vector_type(4)))  __bf16 v4bf;
typedef __attribute__((ext_vector_type(8)))  float  v8f;

static constexpr int T    = 8192;
static constexpr int D    = 1024;
static constexpr int E    = 8;
static constexpr int F    = 2048;
static constexpr int TOPK = 2;

// ---- CDNA5 async global->LDS copy (16B per lane), tracked by ASYNCcnt ----
__device__ __forceinline__ void async_g2l_b128(uint32_t lds_off, const void* g)
{
    asm volatile("global_load_async_to_lds_b128 %0, %1, off"
                 :: "v"(lds_off), "v"(g) : "memory");
}
__device__ __forceinline__ void wait_async0()
{
    asm volatile("s_wait_asynccnt 0x0" ::: "memory");
}

// ---------------------------------------------------------------------------
// Kernel 1: gating.  logits = x @ wg, top-2, softmax over the 2, push
// (token*2+k) entries into per-expert compacted lists.
// ---------------------------------------------------------------------------
__global__ __launch_bounds__(256) void moe_gate(
    const float* __restrict__ x, const float* __restrict__ wg,
    unsigned int* __restrict__ counts, int* __restrict__ lists,
    float* __restrict__ probs)
{
    const int t = blockIdx.x * 256 + threadIdx.x;
    if (t >= T) return;

    float acc[E];
#pragma unroll
    for (int e = 0; e < E; ++e) acc[e] = 0.f;

    const float* xr = x + (size_t)t * D;
    for (int d = 0; d < D; ++d) {
        const float xv = xr[d];
        const float* wr = wg + (size_t)d * E;   // uniform address -> scalar path
#pragma unroll
        for (int e = 0; e < E; ++e) acc[e] = fmaf(xv, wr[e], acc[e]);
    }

    int s0 = 0; float v0 = acc[0];
#pragma unroll
    for (int e = 1; e < E; ++e) if (acc[e] > v0) { v0 = acc[e]; s0 = e; }
    int s1 = -1; float v1 = -3.4e38f;
#pragma unroll
    for (int e = 0; e < E; ++e) if (e != s0 && acc[e] > v1) { v1 = acc[e]; s1 = e; }

    const float e1  = __expf(v1 - v0);
    const float inv = 1.f / (1.f + e1);

    {
        const unsigned int pos = atomicAdd(&counts[s0], 1u);
        lists[(size_t)s0 * T + pos] = t * 2 + 0;
        probs[t * 2 + 0] = inv;
    }
    {
        const unsigned int pos = atomicAdd(&counts[s1], 1u);
        lists[(size_t)s1 * T + pos] = t * 2 + 1;
        probs[t * 2 + 1] = e1 * inv;
    }
}

// ---------------------------------------------------------------------------
// Kernel 2: up-projection.  H = silu(X @ w1) * (X @ w3) per gathered 128-token
// tile x 128-F tile.  Double-buffered LDS; fp32->bf16 staged through regs so
// global loads for chunk k+1 overlap the WMMAs of chunk k.
// ---------------------------------------------------------------------------
__global__ __launch_bounds__(256) void moe_up(
    const float* __restrict__ x,
    const float* __restrict__ w1, const float* __restrict__ w3,
    const unsigned int* __restrict__ counts, const int* __restrict__ lists,
    __bf16* __restrict__ H)
{
    const int e   = blockIdx.z;
    const int m0  = blockIdx.y * 128;
    const int nb0 = blockIdx.x * 128;
    const int cnt = (int)counts[e];
    if (m0 >= cnt) return;

    __shared__ alignas(32) __bf16 lA [2][128 * 32];
    __shared__ alignas(32) __bf16 lB1[2][32 * 128];
    __shared__ alignas(32) __bf16 lB3[2][32 * 128];
    __shared__ int lent[128];

    const int tid = threadIdx.x;
    if (tid < 128) {
        const int gr = m0 + tid;
        lent[tid] = (gr < cnt) ? lists[(size_t)e * T + gr] : 0;
    }
    __syncthreads();

    // ---- loop-invariant staging addresses ----
    const int r0 = tid >> 3;            // A: 32 rows per pass
    const int k4 = (tid & 7) * 4;
    const float* arow[4];
#pragma unroll
    for (int p = 0; p < 4; ++p)
        arow[p] = x + (size_t)(lent[p * 32 + r0] >> 1) * D + k4;

    const int kk0 = tid >> 5;           // B: 8 k-rows per pass
    const int n4  = (tid & 31) * 4;
    const float* w1p = w1 + ((size_t)e * D + kk0) * F + nb0 + n4;
    const float* w3p = w3 + ((size_t)e * D + kk0) * F + nb0 + n4;

    const int  lane = tid & 31;
    const int  wave = tid >> 5;
    const int  wm   = wave >> 2;
    const int  wn   = wave & 3;
    const bool lo   = lane < 16;
    const int  lm   = lane & 15;

    v8f acc1[4][2], acc3[4][2];
    const v8f vzero = {0.f, 0.f, 0.f, 0.f, 0.f, 0.f, 0.f, 0.f};
#pragma unroll
    for (int i = 0; i < 4; ++i)
#pragma unroll
        for (int j = 0; j < 2; ++j) { acc1[i][j] = vzero; acc3[i][j] = vzero; }

    float4 ra[4], rb1[4], rb3[4];

    auto load_regs = [&]() {
#pragma unroll
        for (int p = 0; p < 4; ++p) {
            ra[p]  = *(const float4*)(arow[p]);
            rb1[p] = *(const float4*)(w1p + (size_t)(p * 8) * F);
            rb3[p] = *(const float4*)(w3p + (size_t)(p * 8) * F);
        }
#pragma unroll
        for (int p = 0; p < 4; ++p) arow[p] += 32;
        w1p += (size_t)32 * F;
        w3p += (size_t)32 * F;
    };
    auto store_lds = [&](int buf) {
#pragma unroll
        for (int p = 0; p < 4; ++p) {
            v4bf a; a.x = (__bf16)ra[p].x;  a.y = (__bf16)ra[p].y;
                    a.z = (__bf16)ra[p].z;  a.w = (__bf16)ra[p].w;
            *(v4bf*)&lA[buf][(p * 32 + r0) * 32 + k4] = a;
            v4bf b; b.x = (__bf16)rb1[p].x; b.y = (__bf16)rb1[p].y;
                    b.z = (__bf16)rb1[p].z; b.w = (__bf16)rb1[p].w;
            *(v4bf*)&lB1[buf][(p * 8 + kk0) * 128 + n4] = b;
            v4bf c; c.x = (__bf16)rb3[p].x; c.y = (__bf16)rb3[p].y;
                    c.z = (__bf16)rb3[p].z; c.w = (__bf16)rb3[p].w;
            *(v4bf*)&lB3[buf][(p * 8 + kk0) * 128 + n4] = c;
        }
    };

    // prologue: chunk 0 -> buffer 0
    load_regs();
    store_lds(0);

    const int NC = D / 32;
    for (int chunk = 0; chunk < NC; ++chunk) {
        __syncthreads();
        const int  cur  = chunk & 1;
        const bool more = (chunk + 1 < NC);
        if (more) load_regs();          // global loads issue before the WMMAs

        v16bf bf1[2], bf3[2];
#pragma unroll
        for (int ns = 0; ns < 2; ++ns) {
            const int nb = wn * 32 + ns * 16;
#pragma unroll
            for (int s = 0; s < 16; ++s) {
                const int kb = (s < 8 ? 0 : 8) + (lo ? 0 : 8) + s;
                bf1[ns][s] = lB1[cur][kb * 128 + nb + lm];
                bf3[ns][s] = lB3[cur][kb * 128 + nb + lm];
            }
        }
#pragma unroll
        for (int ms = 0; ms < 4; ++ms) {
            const int r   = wm * 64 + ms * 16 + lm;
            const int off = lo ? 0 : 8;
            const v8bf a0 = *(const v8bf*)&lA[cur][r * 32 + off];
            const v8bf a1 = *(const v8bf*)&lA[cur][r * 32 + 16 + off];
            const v16bf a = __builtin_shufflevector(a0, a1,
                0, 1, 2, 3, 4, 5, 6, 7, 8, 9, 10, 11, 12, 13, 14, 15);
#pragma unroll
            for (int ns = 0; ns < 2; ++ns) {
                acc1[ms][ns] = __builtin_amdgcn_wmma_f32_16x16x32_bf16(
                    false, a, false, bf1[ns], (short)0, acc1[ms][ns], false, false);
                acc3[ms][ns] = __builtin_amdgcn_wmma_f32_16x16x32_bf16(
                    false, a, false, bf3[ns], (short)0, acc3[ms][ns], false, false);
            }
        }
        if (more) store_lds(cur ^ 1);   // wait/convert/store after the WMMAs
    }

    // ---- epilogue: h = silu(a) * b -> H (bf16) ----
#pragma unroll
    for (int ms = 0; ms < 4; ++ms) {
        const int rbase = wm * 64 + ms * 16 + (lo ? 0 : 8);
#pragma unroll
        for (int ns = 0; ns < 2; ++ns) {
            const int col = nb0 + wn * 32 + ns * 16 + lm;
#pragma unroll
            for (int r = 0; r < 8; ++r) {
                const int rl = rbase + r;
                if (m0 + rl < cnt) {
                    const float a = acc1[ms][ns][r];
                    const float b = acc3[ms][ns][r];
                    const float h = (a / (1.f + __expf(-a))) * b;
                    H[(size_t)lent[rl] * F + col] = (__bf16)h;
                }
            }
        }
    }
}

// ---------------------------------------------------------------------------
// Kernel 3: down-projection.  partial = prob * (H @ w2).  The bf16 H tile is
// DMA'd straight into LDS with global_load_async_to_lds_b128 (no conversion
// needed); the fp32 w2 tile goes through registers for the bf16 convert.
// ---------------------------------------------------------------------------
__global__ __launch_bounds__(256) void moe_down(
    const __bf16* __restrict__ H, const float* __restrict__ w2,
    const unsigned int* __restrict__ counts, const int* __restrict__ lists,
    const float* __restrict__ probs, float* __restrict__ partial)
{
    const int e   = blockIdx.z;
    const int m0  = blockIdx.y * 128;
    const int nb0 = blockIdx.x * 128;
    const int cnt = (int)counts[e];
    if (m0 >= cnt) return;

    __shared__ alignas(32) __bf16 lA[2][128 * 32];
    __shared__ alignas(32) __bf16 lB[2][32 * 128];
    __shared__ int   lent [128];
    __shared__ float lprob[128];

    const int tid = threadIdx.x;
    if (tid < 128) {
        const int gr = m0 + tid;
        const int en = (gr < cnt) ? lists[(size_t)e * T + gr] : 0;
        lent[tid]  = en;
        lprob[tid] = (gr < cnt) ? probs[en] : 0.f;
    }
    __syncthreads();

    // ---- async A staging: each thread owns 32B of one H row per chunk ----
    const int arow = tid >> 1;                 // 0..127
    const int aseg = (tid & 1) * 16;           // bf16 elements within the row
    const __bf16* hrow = H + (size_t)lent[arow] * F + aseg;
    const uint32_t loff0 = (uint32_t)(uintptr_t)&lA[0][arow * 32 + aseg];
    const uint32_t loff1 = (uint32_t)(uintptr_t)&lA[1][arow * 32 + aseg];

    auto async_A = [&](int buf) {
        const uint32_t lo_ = buf ? loff1 : loff0;
        async_g2l_b128(lo_,      hrow);        // 16B
        async_g2l_b128(lo_ + 16, hrow + 8);    // 16B
        hrow += 32;
    };

    // ---- register staging for w2 (fp32 -> bf16) ----
    const int kk0 = tid >> 5;
    const int n4  = (tid & 31) * 4;
    const float* w2p = w2 + ((size_t)e * F + kk0) * D + nb0 + n4;
    float4 rb[4];
    auto load_B = [&]() {
#pragma unroll
        for (int p = 0; p < 4; ++p)
            rb[p] = *(const float4*)(w2p + (size_t)(p * 8) * D);
        w2p += (size_t)32 * D;
    };
    auto store_B = [&](int buf) {
#pragma unroll
        for (int p = 0; p < 4; ++p) {
            v4bf b; b.x = (__bf16)rb[p].x; b.y = (__bf16)rb[p].y;
                    b.z = (__bf16)rb[p].z; b.w = (__bf16)rb[p].w;
            *(v4bf*)&lB[buf][(p * 8 + kk0) * 128 + n4] = b;
        }
    };

    const int  lane = tid & 31;
    const int  wave = tid >> 5;
    const int  wm   = wave >> 2;
    const int  wn   = wave & 3;
    const bool lo   = lane < 16;
    const int  lm   = lane & 15;

    v8f acc[4][2];
    const v8f vzero = {0.f, 0.f, 0.f, 0.f, 0.f, 0.f, 0.f, 0.f};
#pragma unroll
    for (int i = 0; i < 4; ++i)
#pragma unroll
        for (int j = 0; j < 2; ++j) acc[i][j] = vzero;

    // prologue: chunk 0 -> buffer 0
    async_A(0);
    load_B();
    store_B(0);
    wait_async0();

    const int NC = F / 32;
    for (int chunk = 0; chunk < NC; ++chunk) {
        __syncthreads();
        const int  cur  = chunk & 1;
        const bool more = (chunk + 1 < NC);
        if (more) {
            async_A(cur ^ 1);           // DMA next H tile during the WMMAs
            load_B();
        }

        v16bf bf[2];
#pragma unroll
        for (int ns = 0; ns < 2; ++ns)
#pragma unroll
            for (int s = 0; s < 16; ++s) {
                const int kb = (s < 8 ? 0 : 8) + (lo ? 0 : 8) + s;
                bf[ns][s] = lB[cur][kb * 128 + wn * 32 + ns * 16 + lm];
            }
#pragma unroll
        for (int ms = 0; ms < 4; ++ms) {
            const int r   = wm * 64 + ms * 16 + lm;
            const int off = lo ? 0 : 8;
            const v8bf a0 = *(const v8bf*)&lA[cur][r * 32 + off];
            const v8bf a1 = *(const v8bf*)&lA[cur][r * 32 + 16 + off];
            const v16bf a = __builtin_shufflevector(a0, a1,
                0, 1, 2, 3, 4, 5, 6, 7, 8, 9, 10, 11, 12, 13, 14, 15);
#pragma unroll
            for (int ns = 0; ns < 2; ++ns)
                acc[ms][ns] = __builtin_amdgcn_wmma_f32_16x16x32_bf16(
                    false, a, false, bf[ns], (short)0, acc[ms][ns], false, false);
        }
        if (more) {
            store_B(cur ^ 1);
            wait_async0();              // our DMA must land before next barrier
        }
    }

    // ---- epilogue: scale by routing prob, scatter to per-slot partial ----
#pragma unroll
    for (int ms = 0; ms < 4; ++ms) {
        const int rbase = wm * 64 + ms * 16 + (lo ? 0 : 8);
#pragma unroll
        for (int ns = 0; ns < 2; ++ns) {
            const int col = nb0 + wn * 32 + ns * 16 + lm;
#pragma unroll
            for (int r = 0; r < 8; ++r) {
                const int rl = rbase + r;
                if (m0 + rl < cnt)
                    partial[(size_t)lent[rl] * D + col] = lprob[rl] * acc[ms][ns][r];
            }
        }
    }
}

// ---------------------------------------------------------------------------
// Kernel 4: out[t] = partial[t,0] + partial[t,1]  (float4 vectorized)
// ---------------------------------------------------------------------------
__global__ __launch_bounds__(256) void moe_combine(
    const float* __restrict__ partial, float* __restrict__ out)
{
    const size_t i  = (size_t)blockIdx.x * 256 + threadIdx.x;
    const size_t N4 = (size_t)T * D / 4;
    if (i >= N4) return;
    const size_t d4 = i % (D / 4);
    const size_t t  = i / (D / 4);
    const float4* p = (const float4*)partial;
    const float4 a = p[(2 * t)     * (D / 4) + d4];
    const float4 b = p[(2 * t + 1) * (D / 4) + d4];
    float4 o;
    o.x = a.x + b.x; o.y = a.y + b.y; o.z = a.z + b.z; o.w = a.w + b.w;
    ((float4*)out)[i] = o;
}

// ---------------------------------------------------------------------------
extern "C" void kernel_launch(void* const* d_in, const int* in_sizes, int n_in,
                              void* d_out, int out_size, void* d_ws, size_t ws_size,
                              hipStream_t stream)
{
    (void)in_sizes; (void)n_in; (void)out_size; (void)ws_size;
    const float* x  = (const float*)d_in[0];
    const float* wg = (const float*)d_in[1];
    const float* w1 = (const float*)d_in[2];
    const float* w3 = (const float*)d_in[3];
    const float* w2 = (const float*)d_in[4];
    float* out = (float*)d_out;

    char* ws = (char*)d_ws;
    const size_t off_lists   = 256;
    const size_t off_probs   = off_lists + (size_t)E * T * sizeof(int);
    const size_t off_H       = off_probs + (size_t)T * TOPK * sizeof(float);
    const size_t off_partial = off_H + (size_t)T * TOPK * F * sizeof(__bf16);

    unsigned int* counts  = (unsigned int*)ws;
    int*          lists   = (int*)   (ws + off_lists);
    float*        probs   = (float*) (ws + off_probs);
    __bf16*       Hbuf    = (__bf16*)(ws + off_H);
    float*        partial = (float*) (ws + off_partial);

    hipMemsetAsync(counts, 0, E * sizeof(unsigned int), stream);

    moe_gate<<<T / 256, 256, 0, stream>>>(x, wg, counts, lists, probs);
    moe_up  <<<dim3(F / 128, T / 128, E), 256, 0, stream>>>(x, w1, w3, counts, lists, Hbuf);
    moe_down<<<dim3(D / 128, T / 128, E), 256, 0, stream>>>(Hbuf, w2, counts, lists, probs, partial);
    moe_combine<<<(int)(((size_t)T * D / 4 + 255) / 256), 256, 0, stream>>>(partial, out);
}
